// Net_10960756540250
// MI455X (gfx1250) — matrix operand order repaired and checked by reading, not compiled, not run on hardware
//
#include <hip/hip_runtime.h>

#define IN_C  602
#define HID_C 256
#define OUT_C 41
#define CLIPV 1000000.0f

typedef __attribute__((ext_vector_type(16))) _Float16 v16h;
typedef __attribute__((ext_vector_type(8)))  _Float16 v8h;
typedef __attribute__((ext_vector_type(8)))  float    v8f;

__device__ __forceinline__ float ntn(float v) {
  if (!__builtin_isfinite(v)) v = (v != v) ? 0.0f : (v > 0.0f ? CLIPV : -CLIPV);
  return v;
}

__device__ __forceinline__ v8f wmma_f16(v16h a, v16h b, v8f c) {
  return __builtin_amdgcn_wmma_f32_16x16x32_f16(false, a, false, b, (short)0, c,
                                                false, false);
}

// ---------------- GEMM1: h1[Mpad,256] = relu(ntn(x) @ W1^T + b1)  (f16 out) --
// Block tile: 64(M) x 128(N) x 32(K). 8 waves = 4(M) x 2(N); wave = 16x64.
__global__ __launch_bounds__(256) void gemm1_kernel(
    const float* __restrict__ x, const float* __restrict__ W1,
    const float* __restrict__ b1, _Float16* __restrict__ h1, int Nrows) {
  __shared__ alignas(16) _Float16 As[64][40];   // [m][k], stride 40 halves (80B)
  __shared__ alignas(16) _Float16 Bs[128][40];  // [n][k]

  const int tid  = threadIdx.x;
  const int lane = tid & 31;
  const int wv   = tid >> 5;   // 0..7
  const int wm   = wv >> 1;    // 0..3 : 16-row slice
  const int wn   = wv & 1;     // 0..1 : 64-col slice
  const int lh   = lane & 15;
  const int sel  = lane >> 4;  // half-wave select

  const int mTile = blockIdx.x * 64;
  const int nTile = blockIdx.y * 128;

  // A loader: 64 rows x 32 cols f32 -> f16, 256 thr * 4 float2
  const int ldRowA = tid >> 2;        // 0..63
  const int ldC0A  = (tid & 3) * 2;   // 0,2,4,6
  // B loader: 128 rows x 32 cols f32 -> f16, 256 thr * 8 float2
  const int ldRowB = tid >> 1;        // 0..127
  const int ldC0B  = (tid & 1) * 2;   // 0,2

  const int  gmA   = mTile + ldRowA;
  const bool mOK   = (gmA < Nrows);
  const float* xrow = x + (size_t)(mOK ? gmA : 0) * IN_C;  // clamped, branchless
  const float* wrow = W1 + (size_t)(nTile + ldRowB) * IN_C;

  v8f acc[4] = {{}, {}, {}, {}};

  for (int kt = 0; kt < 608; kt += 32) {  // 608 = ceil(602/32)*32
    if (kt < 576) {
      // ---- main path: no K guard (gk max = 575 < 602), straight-line loads
      #pragma unroll
      for (int i = 0; i < 4; ++i) {
        const int k = ldC0A + i * 8;
        const float2 f = *(const float2*)(xrow + kt + k);
        As[ldRowA][k]     = (_Float16)ntn(mOK ? f.x : 0.0f);
        As[ldRowA][k + 1] = (_Float16)ntn(mOK ? f.y : 0.0f);
      }
      #pragma unroll
      for (int i = 0; i < 8; ++i) {
        const int k = ldC0B + i * 4;
        const float2 w = *(const float2*)(wrow + kt + k);
        Bs[ldRowB][k]     = (_Float16)w.x;
        Bs[ldRowB][k + 1] = (_Float16)w.y;
      }
    } else {
      // ---- tail step (kt == 576): K guard for 602..607
      #pragma unroll
      for (int i = 0; i < 4; ++i) {
        const int k  = ldC0A + i * 8;
        const int gk = kt + k;
        const bool ok = (gk + 1) < IN_C;
        const float2 f = *(const float2*)(xrow + (ok ? gk : 0));
        As[ldRowA][k]     = (_Float16)ntn((mOK && ok) ? f.x : 0.0f);
        As[ldRowA][k + 1] = (_Float16)ntn((mOK && ok) ? f.y : 0.0f);
      }
      #pragma unroll
      for (int i = 0; i < 8; ++i) {
        const int k  = ldC0B + i * 4;
        const int gk = kt + k;
        const bool ok = (gk + 1) < IN_C;
        const float2 w = *(const float2*)(wrow + (ok ? gk : 0));
        Bs[ldRowB][k]     = (_Float16)(ok ? w.x : 0.0f);
        Bs[ldRowB][k + 1] = (_Float16)(ok ? w.y : 0.0f);
      }
    }
    __syncthreads();

    // A fragment (16x32): two contiguous 16B LDS reads per lane
    const v8h alo = *(const v8h*)&As[wm * 16 + lh][sel * 8];
    const v8h ahi = *(const v8h*)&As[wm * 16 + lh][sel * 8 + 16];
    v16h a;
    #pragma unroll
    for (int i = 0; i < 8; ++i) { a[i] = alo[i]; a[i + 8] = ahi[i]; }

    // 4 B fragments -> 4 WMMAs reusing the same A fragment
    #pragma unroll
    for (int t = 0; t < 4; ++t) {
      const v8h blo = *(const v8h*)&Bs[wn * 64 + t * 16 + lh][sel * 8];
      const v8h bhi = *(const v8h*)&Bs[wn * 64 + t * 16 + lh][sel * 8 + 16];
      v16h bf;
      #pragma unroll
      for (int i = 0; i < 8; ++i) { bf[i] = blo[i]; bf[i + 8] = bhi[i]; }
      acc[t] = wmma_f16(a, bf, acc[t]);
    }
    __syncthreads();
  }

  #pragma unroll
  for (int t = 0; t < 4; ++t) {
    const int n  = nTile + wn * 64 + t * 16 + lh;
    const float bb = b1[n];
    #pragma unroll
    for (int j = 0; j < 8; ++j) {
      const int gm = mTile + wm * 16 + j + sel * 8;  // C VGPR j -> row j (+8)
      h1[(size_t)gm * HID_C + n] = (_Float16)fmaxf(acc[t][j] + bb, 0.0f);
    }
  }
}

// ------- pad/convert W2 -> f16 [48][256] row-major (rows 41..47 zero) -------
__global__ void prep_w2_kernel(const float* __restrict__ W2,
                               _Float16* __restrict__ W2p) {
  const int idx = blockIdx.x * blockDim.x + threadIdx.x;  // 48*256 threads
  const int n = idx >> 8, k = idx & 255;
  const float v = (n < OUT_C) ? W2[(size_t)n * HID_C + k] : 0.0f;
  W2p[idx] = (_Float16)v;
}

// ---- GEMM2 + bias + ntn + row L2-normalize fused: h0[N,41] (f32) -----------
__global__ __launch_bounds__(256) void gemm2_norm_kernel(
    const _Float16* __restrict__ h1, const _Float16* __restrict__ W2p,
    const float* __restrict__ b2, float* __restrict__ h0, int Nrows) {
  const int tid  = threadIdx.x;
  const int lane = tid & 31;
  const int wv   = tid >> 5;
  const int lh   = lane & 15;
  const int sel  = lane >> 4;

  const int mBase = blockIdx.x * 128 + wv * 16;

  v8f c0 = {}, c1 = {}, c2 = {};
  const _Float16* arow = h1 + (size_t)(mBase + lh) * HID_C;
  const _Float16* p0   = W2p + (size_t)lh        * HID_C;
  const _Float16* p1   = W2p + (size_t)(lh + 16) * HID_C;
  const _Float16* p2   = W2p + (size_t)(lh + 32) * HID_C;

  #pragma unroll
  for (int kt = 0; kt < HID_C; kt += 32) {
    const v16h a = *(const v16h*)(arow + kt + sel * 16);
    const v16h bf0 = *(const v16h*)(p0 + kt + sel * 16);
    const v16h bf1 = *(const v16h*)(p1 + kt + sel * 16);
    const v16h bf2 = *(const v16h*)(p2 + kt + sel * 16);
    c0 = wmma_f16(a, bf0, c0);
    c1 = wmma_f16(a, bf1, c1);
    c2 = wmma_f16(a, bf2, c2);
  }

  const float bb0 = b2[lh];
  const float bb1 = b2[lh + 16];
  const float bb2 = (lh + 32 < OUT_C) ? b2[lh + 32] : 0.0f;

  #pragma unroll
  for (int j = 0; j < 8; ++j) {
    const int gm = mBase + j + sel * 8;
    const float v0 = ntn(c0[j] + bb0);
    const float v1 = ntn(c1[j] + bb1);
    const float v2 = (lh + 32 < OUT_C) ? ntn(c2[j] + bb2) : 0.0f;
    float s = v0 * v0 + v1 * v1 + v2 * v2;
    #pragma unroll
    for (int off = 1; off < 16; off <<= 1) s += __shfl_xor(s, off, 16);
    const float inv = 1.0f / fmaxf(sqrtf(s), 1e-12f);
    if (gm < Nrows) {
      float* o = h0 + (size_t)gm * OUT_C;
      o[lh]      = v0 * inv;
      o[lh + 16] = v1 * inv;
      if (lh + 32 < OUT_C) o[lh + 32] = v2 * inv;
    }
  }
}

// ---------------- degree / dinv -------------------------------------------
__global__ void deg_init_kernel(float* __restrict__ deg, int n) {
  const int i = blockIdx.x * blockDim.x + threadIdx.x;
  if (i < n) deg[i] = 1.0f;  // self loop
}
__global__ void deg_accum_kernel(const int* __restrict__ col,
                                 float* __restrict__ deg, int E) {
  const int stride = gridDim.x * blockDim.x;
  for (int e = blockIdx.x * blockDim.x + threadIdx.x; e < E; e += stride)
    atomicAdd(&deg[col[e]], 1.0f);
}
__global__ void deg_rsqrt_kernel(float* __restrict__ deg, int n) {
  const int i = blockIdx.x * blockDim.x + threadIdx.x;
  if (i < n) deg[i] = rsqrtf(deg[i]);
}

// --- per-iteration: next = alpha*h0 + (1-alpha)*dinv_i^2*cur  (self loop) ---
__global__ void prop_self_kernel(const float* __restrict__ h0,
                                 const float* __restrict__ cur,
                                 const float* __restrict__ dinv,
                                 float* __restrict__ nxt, int n) {
  const int total  = n * OUT_C;
  const int stride = gridDim.x * blockDim.x;
  for (int idx = blockIdx.x * blockDim.x + threadIdx.x; idx < total; idx += stride) {
    const int i = idx / OUT_C;
    const float d = dinv[i];
    nxt[idx] = 0.1f * h0[idx] + 0.9f * d * d * cur[idx];
  }
}

// --- edge scatter: wave per edge, lanes cover 41 features (32 + 9) ----------
__global__ __launch_bounds__(256) void prop_edges_kernel(
    const int* __restrict__ row, const int* __restrict__ col,
    const float* __restrict__ dinv, const float* __restrict__ cur,
    float* __restrict__ nxt, int E) {
  const int lane   = threadIdx.x & 31;
  const int wave   = (blockIdx.x * blockDim.x + threadIdx.x) >> 5;
  const int nWaves = (gridDim.x * blockDim.x) >> 5;
  for (int e = wave; e < E; e += nWaves) {
    const int s = row[e];
    const int t = col[e];
    const float w = 0.9f * dinv[s] * dinv[t];
    const float* src = cur + (size_t)s * OUT_C;
    float*       dst = nxt + (size_t)t * OUT_C;
    atomicAdd(&dst[lane], w * src[lane]);              // features 0..31
    if (lane < OUT_C - 32)
      atomicAdd(&dst[32 + lane], w * src[32 + lane]);  // features 32..40
  }
}

__global__ void final_ntn_kernel(float* __restrict__ out, int total) {
  const int stride = gridDim.x * blockDim.x;
  for (int idx = blockIdx.x * blockDim.x + threadIdx.x; idx < total; idx += stride)
    out[idx] = ntn(out[idx]);
}

// ---------------------------------------------------------------------------
extern "C" void kernel_launch(void* const* d_in, const int* in_sizes, int n_in,
                              void* d_out, int out_size, void* d_ws, size_t ws_size,
                              hipStream_t stream) {
  const float* x   = (const float*)d_in[0];
  const int*   ei  = (const int*)d_in[1];
  const float* W1  = (const float*)d_in[2];
  const float* b1  = (const float*)d_in[3];
  const float* W2  = (const float*)d_in[4];
  const float* b2  = (const float*)d_in[5];
  float* outF = (float*)d_out;

  const int N    = in_sizes[0] / IN_C;
  const int E    = in_sizes[1] / 2;
  const int Mpad = ((N + 127) / 128) * 128;

  const int* rowPtr = ei;        // sources
  const int* colPtr = ei + E;    // targets

  // workspace carve-out (256B aligned slices)
  char* ws = (char*)d_ws;
  size_t off = 0;
  auto take = [&](size_t bytes) -> char* {
    char* p = ws + off;
    off += (bytes + 255) & ~(size_t)255;
    return p;
  };
  _Float16* W2p  = (_Float16*)take((size_t)48 * HID_C * sizeof(_Float16));
  _Float16* h1   = (_Float16*)take((size_t)Mpad * HID_C * sizeof(_Float16));
  float*    h0   = (float*)take((size_t)N * OUT_C * sizeof(float));
  float*    tmp  = (float*)take((size_t)N * OUT_C * sizeof(float));
  float*    dinv = (float*)take((size_t)N * sizeof(float));

  // 1) MLP encoder (WMMA f16, f32 accumulate)
  prep_w2_kernel<<<48, 256, 0, stream>>>(W2, W2p);
  gemm1_kernel<<<dim3(Mpad / 64, HID_C / 128), 256, 0, stream>>>(x, W1, b1, h1, N);
  gemm2_norm_kernel<<<Mpad / 128, 256, 0, stream>>>(h1, W2p, b2, h0, N);

  // 2) GCN normalization terms
  deg_init_kernel<<<(N + 255) / 256, 256, 0, stream>>>(dinv, N);
  deg_accum_kernel<<<2048, 256, 0, stream>>>(colPtr, dinv, E);
  deg_rsqrt_kernel<<<(N + 255) / 256, 256, 0, stream>>>(dinv, N);

  // 3) APPNP: K=10 ping-pong rounds; final round lands in d_out
  const float* cur = h0;
  for (int k = 0; k < 10; ++k) {
    float* nxt = (k & 1) ? outF : tmp;  // k=9 (odd) -> d_out
    prop_self_kernel<<<2048, 256, 0, stream>>>(h0, cur, dinv, nxt, N);
    prop_edges_kernel<<<4096, 256, 0, stream>>>(rowPtr, colPtr, dinv, cur, nxt, E);
    cur = nxt;
  }
  final_ntn_kernel<<<2048, 256, 0, stream>>>(outF, N * OUT_C);
}
